// AFGSA_23699629540061
// MI455X (gfx1250) — matrix operand order repaired and checked by reading, compiled, not run on hardware
//
#include <hip/hip_runtime.h>
#include <stdint.h>

#define PIX 40000
#define HH 200
#define WW 200
#define NBH 40
#define KWIN 11
#define BLK 5

typedef __attribute__((ext_vector_type(16))) __bf16 v16bf;
typedef __attribute__((ext_vector_type(8)))  float  v8f;
typedef __attribute__((ext_vector_type(4)))  unsigned int u32x4;

union Frag { v16bf v; u32x4 q[2]; unsigned u[8]; __bf16 h[16]; };
union Acc  { v8f v; float f[8]; };
union Pk2  { unsigned u; __bf16 h[2]; };

__device__ inline unsigned pack2(float a, float b) {
  Pk2 p; p.h[0] = (__bf16)a; p.h[1] = (__bf16)b; return p.u;
}

__device__ inline v8f wmma_bf16(v16bf a, v16bf b, v8f c) {
  // D = A*B + C, 16x16x32 bf16 -> f32
  return __builtin_amdgcn_wmma_f32_16x16x32_bf16(false, a, false, b, (short)0, c, false, false);
}

// async copy: 16 bytes global -> LDS, tracked by ASYNCcnt, EXEC-masked per lane
__device__ inline void async_b128(unsigned lds_off, unsigned long long gaddr) {
  asm volatile("global_load_async_to_lds_b128 %0, %1, off"
               :: "v"(lds_off), "v"(gaddr) : "memory");
}
__device__ inline unsigned lds_off32(const void* p) {
  return (unsigned)(uintptr_t)p;   // low 32 bits of flat shared addr = LDS offset
}

// ---------------------------------------------------------------------------
// Kernel 0: convert weights f32->bf16 (L2-resident for the GEMM kernel) and
// build rel table relt[j=ki*11+kj][d]: d<32 -> rel_h[ki][d], d>=32 -> rel_w[kj][d-32]
// ---------------------------------------------------------------------------
__global__ __launch_bounds__(256) void prep_kernel(
    const float* __restrict__ Wm, const float* __restrict__ Wq,
    const float* __restrict__ Wk, const float* __restrict__ Wv,
    const float* __restrict__ rel_h, const float* __restrict__ rel_w,
    __bf16* __restrict__ wmb, __bf16* __restrict__ wqb,
    __bf16* __restrict__ wkb, __bf16* __restrict__ wvb,
    __bf16* __restrict__ relt)
{
  int i = blockIdx.x * 256 + threadIdx.x;
  if (i < 131072) wmb[i] = (__bf16)Wm[i];
  if (i < 65536) {
    wqb[i] = (__bf16)Wq[i];
    wkb[i] = (__bf16)Wk[i];
    wvb[i] = (__bf16)Wv[i];
  }
  if (i < 128 * 64) {
    int j = i >> 6, d = i & 63;
    float v = 0.f;
    if (j < 121) {
      int ki = j / KWIN, kj = j - ki * KWIN;
      v = (d < 32) ? rel_h[ki * 32 + d] : rel_w[kj * 32 + (d - 32)];
    }
    relt[i] = (__bf16)v;
  }
}

// ---------------------------------------------------------------------------
// Kernel 1: fused 1x1-conv GEMMs.
//   n_aux = relu(Wmap @ [noisy;aux] + b)   (kept in LDS only)
//   q = 0.125 * Wq @ n_aux ; k = Wk @ n_aux ; v = Wv @ noisy
// outputs bf16, layout [b][head][pixel][64] (d contiguous).
// ---------------------------------------------------------------------------
__global__ __launch_bounds__(256) void qkv_kernel(
    const float* __restrict__ noisy, const float* __restrict__ aux,
    const float* __restrict__ bmap,
    const __bf16* __restrict__ Wm, const __bf16* __restrict__ Wq,
    const __bf16* __restrict__ Wk, const __bf16* __restrict__ Wv,
    __bf16* __restrict__ qw, __bf16* __restrict__ kw, __bf16* __restrict__ vw)
{
  constexpr int NP = 32;
  __shared__ __align__(16) unsigned xl[256 * NP];   // 512 ch of x, packed pairs
  __shared__ __align__(16) unsigned nl[128 * NP];   // 256 ch of n_aux, packed pairs
  const int t = threadIdx.x;
  const int p0g = blockIdx.x * NP;
  const int b = p0g / PIX;
  const int pin0 = p0g - b * PIX;
  const float* nb = noisy + (size_t)b * 256 * PIX;
  const float* ab = aux   + (size_t)b * 256 * PIX;

  #pragma unroll
  for (int it = 0; it < 32; ++it) {                 // 256 cpairs * 32 px / 256 thr
    int idx = t + it * 256;
    int p = idx & (NP - 1);
    int cp = idx >> 5;
    int c = cp * 2;
    const float* src = (c < 256) ? (nb + (size_t)c * PIX)
                                 : (ab + (size_t)(c - 256) * PIX);
    float f0 = src[pin0 + p];
    float f1 = src[PIX + pin0 + p];
    xl[cp * NP + p] = pack2(f0, f1);
  }
  __syncthreads();

  const int wv_ = t >> 5;
  const int lane = t & 31;
  const int hf = lane >> 4;       // half-wave: selects contiguous K half, k = 16*hf + s
  const int l = lane & 15;
  const int mbase = wv_ * 32;

  // ---- map stage: n_aux = relu(Wmap @ x + b) ----
  Acc acc[2][2] = {};
  for (int kk = 0; kk < 16; ++kk) {
    const int k0 = kk * 32;
    Frag a[2], bfr[2];
    #pragma unroll
    for (int mt = 0; mt < 2; ++mt) {
      const __bf16* wp = Wm + (size_t)(mbase + mt * 16 + l) * 512 + k0 + hf * 16;
      a[mt].q[0] = *(const u32x4*)(wp);
      a[mt].q[1] = *(const u32x4*)(wp + 8);
      __builtin_prefetch(wp + 32, 0, 3);            // next K-chunk of this row
    }
    #pragma unroll
    for (int nt = 0; nt < 2; ++nt) {
      const int j = nt * 16 + l;
      const int cpb = (k0 >> 1) + hf * 8;
      #pragma unroll
      for (int u = 0; u < 8; ++u) bfr[nt].u[u] = xl[(cpb + u) * NP + j];
    }
    #pragma unroll
    for (int mt = 0; mt < 2; ++mt)
      #pragma unroll
      for (int nt = 0; nt < 2; ++nt)
        acc[mt][nt].v = wmma_bf16(a[mt].v, bfr[nt].v, acc[mt][nt].v);
  }
  // bias + relu + repack into LDS (C layout: f[r] -> row M = r + 8*hf)
  #pragma unroll
  for (int mt = 0; mt < 2; ++mt) {
    float bias[8];
    #pragma unroll
    for (int r = 0; r < 8; ++r) bias[r] = bmap[mbase + mt * 16 + hf * 8 + r];
    #pragma unroll
    for (int nt = 0; nt < 2; ++nt) {
      #pragma unroll
      for (int r = 0; r < 8; r += 2) {
        float v0 = fmaxf(acc[mt][nt].f[r]     + bias[r],     0.f);
        float v1 = fmaxf(acc[mt][nt].f[r + 1] + bias[r + 1], 0.f);
        nl[((mbase + mt * 16 + hf * 8 + r) >> 1) * NP + nt * 16 + l] = pack2(v0, v1);
      }
    }
  }
  __syncthreads();

  // ---- q/k/v stage (K = 256) ----
  #pragma unroll 1
  for (int o = 0; o < 3; ++o) {
    const __bf16* W = (o == 0) ? Wq : (o == 1) ? Wk : Wv;
    const unsigned* Bsrc = (o < 2) ? nl : xl;       // v uses noisy (cpairs 0..127 of xl)
    Acc oa[2][2] = {};
    for (int kk = 0; kk < 8; ++kk) {
      const int k0 = kk * 32;
      Frag a[2], bfr[2];
      #pragma unroll
      for (int mt = 0; mt < 2; ++mt) {
        const __bf16* wp = W + (size_t)(mbase + mt * 16 + l) * 256 + k0 + hf * 16;
        a[mt].q[0] = *(const u32x4*)(wp);
        a[mt].q[1] = *(const u32x4*)(wp + 8);
      }
      #pragma unroll
      for (int nt = 0; nt < 2; ++nt) {
        const int j = nt * 16 + l;
        const int cpb = (k0 >> 1) + hf * 8;
        #pragma unroll
        for (int u = 0; u < 8; ++u) bfr[nt].u[u] = Bsrc[(cpb + u) * NP + j];
      }
      #pragma unroll
      for (int mt = 0; mt < 2; ++mt)
        #pragma unroll
        for (int nt = 0; nt < 2; ++nt)
          oa[mt][nt].v = wmma_bf16(a[mt].v, bfr[nt].v, oa[mt][nt].v);
    }
    const float sc = (o == 0) ? 0.125f : 1.0f;      // q pre-scaled by D^-0.5
    __bf16* dst = (o == 0) ? qw : (o == 1) ? kw : vw;
    #pragma unroll
    for (int mt = 0; mt < 2; ++mt) {
      const int mtile = wv_ * 2 + mt;
      const int head = mtile >> 2;
      const int d0 = (mtile & 3) * 16 + hf * 8;
      #pragma unroll
      for (int nt = 0; nt < 2; ++nt) {
        const int pin = pin0 + nt * 16 + l;
        unsigned pk[4];
        #pragma unroll
        for (int r = 0; r < 8; r += 2)
          pk[r >> 1] = pack2(oa[mt][nt].f[r] * sc, oa[mt][nt].f[r + 1] * sc);
        *(u32x4*)(dst + ((size_t)(b * 4 + head) * PIX + pin) * 64 + d0) =
            *(const u32x4*)pk;
      }
    }
  }
}

// ---------------------------------------------------------------------------
// Kernel 2: halo attention. One wave per (b,bh,bw,head); 12800 workgroups.
//   sim  = q[25(->32) x 64] @ kp[121(->128) x 64]^T  (kp = k + relt, zero-padded)
//   attn = softmax(sim) ; out = attn @ v
// q/k tiles arrive via GLOBAL_LOAD_ASYNC_TO_LDS_B128 (ASYNCcnt), overlapped
// with the v transpose gather; halo/pad rows come from the DS zero-fill.
// Row stride 72 bf16 => 36-dword lane skew, conflict-free 16B-aligned b128 reads.
// ---------------------------------------------------------------------------
__global__ __launch_bounds__(32) void attn_kernel(
    const __bf16* __restrict__ qw, const __bf16* __restrict__ kw,
    const __bf16* __restrict__ vw, const __bf16* __restrict__ relt,
    float* __restrict__ out)
{
  __shared__ __align__(16) __bf16 q_lds[32 * 72];
  __shared__ __align__(16) __bf16 k_lds[128 * 72];
  __shared__ __align__(16) __bf16 vt[64 * 136];
  __shared__ __align__(16) __bf16 at[32 * 136];
  const int wg = blockIdx.x;
  const int head = wg & 3;
  int tmp = wg >> 2;
  const int bw = tmp % NBH; tmp /= NBH;
  const int bh = tmp % NBH;
  const int b  = tmp / NBH;
  const int lane = threadIdx.x;
  const int hf = lane >> 4;
  const int l = lane & 15;
  const size_t hb = (size_t)(b * 4 + head) * PIX * 64;
  const __bf16* qh = qw + hb;
  const __bf16* kh = kw + hb;
  const __bf16* vh = vw + hb;
  const int y0 = bh * BLK - 3;
  const int x0 = bw * BLK - 3;

  // 1) zero-fill q/k tiles (implements zero padding for halo & pad rows)
  {
    u32x4 z = {0, 0, 0, 0};
    u32x4* qz = (u32x4*)q_lds;
    #pragma unroll
    for (int it = 0; it < 9; ++it) qz[it * 32 + lane] = z;      // 32*72*2 = 9*512B
    u32x4* kz = (u32x4*)k_lds;
    #pragma unroll
    for (int it = 0; it < 36; ++it) kz[it * 32 + lane] = z;     // 128*72*2 = 36*512B
  }
  asm volatile("s_wait_dscnt 0" ::: "memory");  // zeros visible before async lands

  // 2) async q rows (25 x 128B): lane -> (row i, 16B chunk c)
  #pragma unroll
  for (int it = 0; it < 8; ++it) {
    int idx = it * 32 + lane;
    int i = idx >> 3, c = idx & 7;
    if (i < 25) {
      int py = bh * BLK + i / 5, px = bw * BLK + i % 5;
      async_b128(lds_off32(q_lds + i * 72 + c * 8),
                 (unsigned long long)(uintptr_t)(qh + (size_t)(py * WW + px) * 64 + c * 8));
    }
  }
  // 3) async k rows (<=121 x 128B), skipping out-of-image halo rows
  for (int it = 0; it < 32; ++it) {
    int idx = it * 32 + lane;
    int j = idx >> 3, c = idx & 7;
    int ki = j / KWIN, kj = j - ki * KWIN;
    int hy = y0 + ki, hx = x0 + kj;
    if (j < 121 && hy >= 0 && hy < HH && hx >= 0 && hx < WW) {
      async_b128(lds_off32(k_lds + j * 72 + c * 8),
                 (unsigned long long)(uintptr_t)(kh + (size_t)(hy * WW + hx) * 64 + c * 8));
    }
  }

  // 4) foreground: gather v (zero-padded) transposed into vt; d = 2*lane, 2*lane+1
  for (int j = 0; j < 128; ++j) {
    int ki = j / KWIN, kj = j - ki * KWIN;
    int hy = y0 + ki, hx = x0 + kj;
    unsigned u = 0;
    if (j < 121 && hy >= 0 && hy < HH && hx >= 0 && hx < WW)
      u = *(const unsigned*)(vh + (size_t)(hy * WW + hx) * 64 + lane * 2);
    Pk2 p; p.u = u;
    vt[(2 * lane) * 136 + j]     = p.h[0];
    vt[(2 * lane + 1) * 136 + j] = p.h[1];
  }
  asm volatile("s_wait_asynccnt 0" ::: "memory");  // q/k tiles resident
  __syncthreads();

  // ---- sim = q @ kp^T ----
  Acc S[2][8] = {};
  #pragma unroll 1
  for (int kk = 0; kk < 2; ++kk) {
    const int k0 = kk * 32;
    Frag a[2];
    #pragma unroll
    for (int mt = 0; mt < 2; ++mt) {
      const __bf16* ap = q_lds + (mt * 16 + l) * 72 + k0 + hf * 16;
      a[mt].q[0] = *(const u32x4*)ap;
      a[mt].q[1] = *(const u32x4*)(ap + 8);
    }
    #pragma unroll
    for (int nt = 0; nt < 8; ++nt) {
      const int j = nt * 16 + l;
      Frag bfr;
      const __bf16* kp = k_lds + j * 72 + k0 + hf * 16;
      bfr.q[0] = *(const u32x4*)kp;
      bfr.q[1] = *(const u32x4*)(kp + 8);
      if (j < 121) {                      // rel added even on zero-padded pixels
        const __bf16* rp = relt + j * 64 + k0 + hf * 16;
        Frag rf;
        rf.q[0] = *(const u32x4*)rp;
        rf.q[1] = *(const u32x4*)(rp + 8);
        bfr.v = bfr.v + rf.v;             // packed bf16 adds (v_pk_add_bf16)
      }
      #pragma unroll
      for (int mt = 0; mt < 2; ++mt)
        S[mt][nt].v = wmma_bf16(a[mt].v, bfr.v, S[mt][nt].v);
    }
  }

  // mask pad columns j>=121 (tile 7, lanes l>=9)
  if (l >= 9) {
    #pragma unroll
    for (int mt = 0; mt < 2; ++mt)
      #pragma unroll
      for (int r = 0; r < 8; ++r) S[mt][7].f[r] = -1e30f;
  }

  // ---- softmax over j (row i = 16*mt + 8*hf + r lives in 16-lane half) ----
  #pragma unroll
  for (int mt = 0; mt < 2; ++mt) {
    #pragma unroll
    for (int r = 0; r < 8; ++r) {
      float m = -1e30f;
      #pragma unroll
      for (int nt = 0; nt < 8; ++nt) m = fmaxf(m, S[mt][nt].f[r]);
      #pragma unroll
      for (int off = 1; off < 16; off <<= 1) m = fmaxf(m, __shfl_xor(m, off, 32));
      float s = 0.f;
      #pragma unroll
      for (int nt = 0; nt < 8; ++nt) {
        float e = __expf(S[mt][nt].f[r] - m);
        S[mt][nt].f[r] = e;
        s += e;
      }
      #pragma unroll
      for (int off = 1; off < 16; off <<= 1) s += __shfl_xor(s, off, 32);
      const float is = 1.f / s;
      #pragma unroll
      for (int nt = 0; nt < 8; ++nt) S[mt][nt].f[r] *= is;
    }
  }

  // C layout -> LDS [i][j] (bf16) for the A fragments of the PV GEMM
  #pragma unroll
  for (int mt = 0; mt < 2; ++mt)
    #pragma unroll
    for (int nt = 0; nt < 8; ++nt)
      #pragma unroll
      for (int r = 0; r < 8; ++r)
        at[(mt * 16 + hf * 8 + r) * 136 + nt * 16 + l] = (__bf16)S[mt][nt].f[r];
  __syncthreads();

  // ---- out = attn @ v ----
  Acc O[2][4] = {};
  #pragma unroll 1
  for (int kk = 0; kk < 4; ++kk) {
    const int k0 = kk * 32;
    Frag a[2];
    #pragma unroll
    for (int mt = 0; mt < 2; ++mt) {
      const __bf16* ap = at + (mt * 16 + l) * 136 + k0 + hf * 16;
      a[mt].q[0] = *(const u32x4*)ap;
      a[mt].q[1] = *(const u32x4*)(ap + 8);
    }
    #pragma unroll
    for (int nt = 0; nt < 4; ++nt) {
      const __bf16* vp = vt + (nt * 16 + l) * 136 + k0 + hf * 16;
      Frag bfr;
      bfr.q[0] = *(const u32x4*)vp;
      bfr.q[1] = *(const u32x4*)(vp + 8);
      #pragma unroll
      for (int mt = 0; mt < 2; ++mt)
        O[mt][nt].v = wmma_bf16(a[mt].v, bfr.v, O[mt][nt].v);
    }
  }

  // scatter f32 output [b][c][h][w]; L2 (192MB) absorbs the write pattern
  #pragma unroll
  for (int mt = 0; mt < 2; ++mt)
    #pragma unroll
    for (int nt = 0; nt < 4; ++nt)
      #pragma unroll
      for (int r = 0; r < 8; ++r) {
        const int i = mt * 16 + hf * 8 + r;
        if (i < 25) {
          const int py = bh * BLK + i / 5, px = bw * BLK + i % 5;
          const int c = head * 64 + nt * 16 + l;
          out[((size_t)(b * 256 + c) * HH + py) * WW + px] = O[mt][nt].f[r];
        }
      }
}

// ---------------------------------------------------------------------------
extern "C" void kernel_launch(void* const* d_in, const int* in_sizes, int n_in,
                              void* d_out, int out_size, void* d_ws, size_t ws_size,
                              hipStream_t stream) {
  const float* noisy = (const float*)d_in[0];
  const float* auxp  = (const float*)d_in[1];
  const float* W_map = (const float*)d_in[2];
  const float* b_map = (const float*)d_in[3];
  const float* Wq    = (const float*)d_in[4];
  const float* Wk    = (const float*)d_in[5];
  const float* Wv    = (const float*)d_in[6];
  const float* rel_h = (const float*)d_in[7];
  const float* rel_w = (const float*)d_in[8];

  char* ws = (char*)d_ws;
  __bf16* wmb  = (__bf16*)(ws);                       // 256*512 bf16
  __bf16* wqb  = (__bf16*)(ws + 262144);              // 256*256 bf16
  __bf16* wkb  = (__bf16*)(ws + 393216);
  __bf16* wvb  = (__bf16*)(ws + 524288);
  __bf16* relt = (__bf16*)(ws + 655360);              // 128*64 bf16
  const size_t QKV_BYTES = (size_t)2 * 4 * PIX * 64 * 2;   // 40,960,000
  __bf16* qws = (__bf16*)(ws + 671744);
  __bf16* kws = (__bf16*)(ws + 671744 + QKV_BYTES);
  __bf16* vws = (__bf16*)(ws + 671744 + 2 * QKV_BYTES);
  float* outp = (float*)d_out;

  prep_kernel<<<512, 256, 0, stream>>>(W_map, Wq, Wk, Wv, rel_h, rel_w,
                                       wmb, wqb, wkb, wvb, relt);
  qkv_kernel<<<(2 * PIX) / 32, 256, 0, stream>>>(noisy, auxp, b_map,
                                                 wmb, wqb, wkb, wvb,
                                                 qws, kws, vws);
  attn_kernel<<<2 * NBH * NBH * 4, 32, 0, stream>>>(qws, kws, vws, relt, outp);
}
